// SeerBlock_76647986365065
// MI455X (gfx1250) — compile-verified, matrix operand order checked by reading, not run on hardware
//
#include <hip/hip_runtime.h>

typedef __bf16 bf16_t;
typedef __bf16 v16bf __attribute__((ext_vector_type(16)));
typedef float  v8f   __attribute__((ext_vector_type(8)));
typedef unsigned int u32x4 __attribute__((ext_vector_type(4)));

#define N_V 50000
#define N_E 800000

static __device__ __forceinline__ v8f wmma_bf16(v16bf a, v16bf b, v8f c) {
    return __builtin_amdgcn_wmma_f32_16x16x32_bf16(false, a, false, b, (short)0, c, false, false);
}

// A-fragment for 16-bit A 16x32: lane L (m = L&15, half = L>>4):
//   elems 0..7  = K(ktBase + 8*half + j)
//   elems 8..15 = K(ktBase + 16 + 8*half + j)
// -> two aligned 16B chunks at halfword offsets off and off+16.
static __device__ __forceinline__ v16bf lds_afrag(const bf16_t* p, int offHalves) {
    union { v16bf v; u32x4 q[2]; } u;
    u.q[0] = *reinterpret_cast<const u32x4*>(p + offHalves);
    u.q[1] = *reinterpret_cast<const u32x4*>(p + offHalves + 16);
    return u.v;
}

static __device__ __forceinline__ int f32_to_key(float x) {
    int b = __float_as_int(x);
    return b ^ ((b >> 31) & 0x7FFFFFFF);   // monotone float->int map
}
static __device__ __forceinline__ float key_to_f32(int k) {
    int b = k ^ ((k >> 31) & 0x7FFFFFFF);
    return __int_as_float(b);
}

// ---------------------------------------------------------------- init ws
__global__ __launch_bounds__(256) void init_ws_kernel(float* __restrict__ agg,
                                                      float* __restrict__ deg,
                                                      float* __restrict__ vsum,
                                                      int* __restrict__ vmaxk) {
    size_t i = (size_t)blockIdx.x * 256 + threadIdx.x;
    if (i < (size_t)N_V * 128) agg[i] = 0.0f;
    if (i < N_V) deg[i] = 0.0f;
    if (i < 128) { vsum[i] = 0.0f; vmaxk[i] = (int)0x80000000; }
}

// ------------------------------------------------- pack W[K][N] -> B fragments
// Fragment (kt,nt) = 32 lanes x 16 bf16 contiguous. Lane L: n = nt*16 + (L&15),
// half = L>>4; elem j: k = kt*32 + (j<8 ? 8*half + j : 16 + 8*half + (j-8)).
__global__ __launch_bounds__(256) void pack_w_kernel(const float* __restrict__ W,
                                                     bf16_t* __restrict__ dst,
                                                     int K, int N) {
    int f = blockIdx.x * 256 + threadIdx.x;
    if (f >= K * N) return;
    int NT = N >> 4;
    int frag = f >> 9;          // /512 halves per fragment
    int r = f & 511;
    int lane = r >> 4;
    int j = r & 15;
    int kt = frag / NT, nt = frag - kt * NT;
    int n = nt * 16 + (lane & 15);
    int half = lane >> 4;
    int k = kt * 32 + ((j < 8) ? (8 * half + j) : (16 + 8 * half + (j - 8)));
    dst[f] = (bf16_t)W[(size_t)k * N + n];
}

// ---------------------------------------------------------------- edge MLP
// 64 edges/block, 256 threads (8 waves). Wave w: mtile = w>>1, ngrp = w&1.
__global__ __launch_bounds__(256) void edge_kernel(
    const float* __restrict__ E, const float* __restrict__ V,
    const int* __restrict__ srcI, const int* __restrict__ dstI,
    const bf16_t* __restrict__ W1p, const float* __restrict__ b1,
    const bf16_t* __restrict__ W2p, const float* __restrict__ b2,
    float* __restrict__ agg, float* __restrict__ deg)
{
    __shared__ __align__(32) bf16_t sX[64 * 336];   // X (stride 336) then H (stride 136)
    __shared__ int sSrc[64], sDst[64];

    const int tid = threadIdx.x;
    const int e0 = blockIdx.x * 64;

    if (tid < 64) {
        int s = srcI[e0 + tid], d = dstI[e0 + tid];
        sSrc[tid] = s; sDst[tid] = d;
        atomicAdd(&deg[d], 1.0f);
    }
    __syncthreads();

    // stage concat[E, V_src, V_dst] as bf16
    for (int i = tid; i < 64 * 320; i += 256) {
        int r = i / 320, c = i - r * 320;
        float v;
        if (c < 64)       v = E[(size_t)(e0 + r) * 64 + c];
        else if (c < 192) v = V[(size_t)sSrc[r] * 128 + (c - 64)];
        else              v = V[(size_t)sDst[r] * 128 + (c - 192)];
        sX[r * 336 + c] = (bf16_t)v;
    }
    __syncthreads();

    const int lane = tid & 31, wave = tid >> 5;
    const int mtile = wave >> 1, ngrp = wave & 1;
    const int hh = lane >> 4, ml = lane & 15;
    const bf16_t* arow = sX + (16 * mtile + ml) * 336;

    const v8f vz = {0,0,0,0,0,0,0,0};
    v8f acc[4] = {vz, vz, vz, vz};

    const v16bf* B1 = (const v16bf*)W1p;
    for (int kt = 0; kt < 10; ++kt) {               // K = 320
        v16bf a = lds_afrag(arow, kt * 32 + 8 * hh);
        const v16bf* bb = B1 + (size_t)(kt * 8 + 4 * ngrp) * 32 + lane;
        #pragma unroll
        for (int t = 0; t < 4; ++t) acc[t] = wmma_bf16(a, bb[t * 32], acc[t]);
    }
    __syncthreads();                                // all waves done reading X

    bf16_t* sH = sX;                                // alias H over X
    #pragma unroll
    for (int t = 0; t < 4; ++t) {
        int col = 64 * ngrp + 16 * t + ml;
        float bias = b1[col];
        #pragma unroll
        for (int r = 0; r < 8; ++r) {
            float hv = acc[t][r] + bias;
            sH[(16 * mtile + r + 8 * hh) * 136 + col] = (bf16_t)fmaxf(hv, 0.0f);
        }
    }
    __syncthreads();

    v8f acc2[4] = {vz, vz, vz, vz};
    const bf16_t* arow2 = sH + (16 * mtile + ml) * 136;
    const v16bf* B2 = (const v16bf*)W2p;
    for (int kt = 0; kt < 4; ++kt) {                // K = 128
        v16bf a = lds_afrag(arow2, kt * 32 + 8 * hh);
        const v16bf* bb = B2 + (size_t)(kt * 8 + 4 * ngrp) * 32 + lane;
        #pragma unroll
        for (int t = 0; t < 4; ++t) acc2[t] = wmma_bf16(a, bb[t * 32], acc2[t]);
    }

    // fused scatter-add into edge_agg (mean normalization happens in node kernel)
    #pragma unroll
    for (int t = 0; t < 4; ++t) {
        int col = 64 * ngrp + 16 * t + ml;
        float bias = b2[col];
        #pragma unroll
        for (int r = 0; r < 8; ++r) {
            int row = 16 * mtile + r + 8 * hh;
            atomicAdd(&agg[(size_t)sDst[row] * 128 + col], acc2[t][r] + bias);
        }
    }
}

// ---------------------------------------------------------------- node MLP
__global__ __launch_bounds__(256) void node_kernel(
    const float* __restrict__ V, const float* __restrict__ agg,
    const float* __restrict__ deg, const float* __restrict__ z,
    const float* __restrict__ u,
    const bf16_t* __restrict__ W1p, const float* __restrict__ b1,
    const bf16_t* __restrict__ W2p, const float* __restrict__ b2,
    float* __restrict__ Vout, float* __restrict__ vsum, int* __restrict__ vmaxk)
{
    __shared__ __align__(32) bf16_t sX[64 * 456];   // X (stride 456) then H (stride 136)
    __shared__ float sDegInv[64];
    __shared__ float sSum[128];
    __shared__ int   sMax[128];

    const int tid = threadIdx.x;
    const int v0 = blockIdx.x * 64;

    if (tid < 64) {
        float d = (v0 + tid < N_V) ? deg[v0 + tid] : 1.0f;
        sDegInv[tid] = 1.0f / fmaxf(d, 1.0f);
    }
    if (tid < 128) { sSum[tid] = 0.0f; sMax[tid] = (int)0x80000000; }
    __syncthreads();

    // stage concat[edge_agg/deg, V, z, u] as bf16 (zero pad past N_V)
    for (int i = tid; i < 64 * 448; i += 256) {
        int r = i / 448, c = i - r * 448;
        int v = v0 + r;
        float val = 0.0f;
        if (v < N_V) {
            if (c < 128)      val = agg[(size_t)v * 128 + c] * sDegInv[r];
            else if (c < 256) val = V[(size_t)v * 128 + (c - 128)];
            else if (c < 384) val = z[c - 256];
            else              val = u[c - 384];
        }
        sX[r * 456 + c] = (bf16_t)val;
    }
    __syncthreads();

    const int lane = tid & 31, wave = tid >> 5;
    const int mtile = wave >> 1, ngrp = wave & 1;
    const int hh = lane >> 4, ml = lane & 15;
    const bf16_t* arow = sX + (16 * mtile + ml) * 456;

    const v8f vz = {0,0,0,0,0,0,0,0};
    v8f acc[4] = {vz, vz, vz, vz};

    const v16bf* B1 = (const v16bf*)W1p;
    for (int kt = 0; kt < 14; ++kt) {               // K = 448
        v16bf a = lds_afrag(arow, kt * 32 + 8 * hh);
        const v16bf* bb = B1 + (size_t)(kt * 8 + 4 * ngrp) * 32 + lane;
        #pragma unroll
        for (int t = 0; t < 4; ++t) acc[t] = wmma_bf16(a, bb[t * 32], acc[t]);
    }
    __syncthreads();

    bf16_t* sH = sX;
    #pragma unroll
    for (int t = 0; t < 4; ++t) {
        int col = 64 * ngrp + 16 * t + ml;
        float bias = b1[col];
        #pragma unroll
        for (int r = 0; r < 8; ++r) {
            float hv = acc[t][r] + bias;
            sH[(16 * mtile + r + 8 * hh) * 136 + col] = (bf16_t)fmaxf(hv, 0.0f);
        }
    }
    __syncthreads();

    v8f acc2[4] = {vz, vz, vz, vz};
    const bf16_t* arow2 = sH + (16 * mtile + ml) * 136;
    const v16bf* B2 = (const v16bf*)W2p;
    for (int kt = 0; kt < 4; ++kt) {                // K = 128
        v16bf a = lds_afrag(arow2, kt * 32 + 8 * hh);
        const v16bf* bb = B2 + (size_t)(kt * 8 + 4 * ngrp) * 32 + lane;
        #pragma unroll
        for (int t = 0; t < 4; ++t) acc2[t] = wmma_bf16(a, bb[t * 32], acc2[t]);
    }

    // store V' + block-local column sum/max, then one global atomic per feature
    #pragma unroll
    for (int t = 0; t < 4; ++t) {
        int col = 64 * ngrp + 16 * t + ml;
        float bias = b2[col];
        float lsum = 0.0f, lmax = -__builtin_inff();
        #pragma unroll
        for (int r = 0; r < 8; ++r) {
            int row = v0 + 16 * mtile + r + 8 * hh;
            float val = acc2[t][r] + bias;
            if (row < N_V) {
                Vout[(size_t)row * 128 + col] = val;
                lsum += val;
                lmax = fmaxf(lmax, val);
            }
        }
        atomicAdd(&sSum[col], lsum);
        atomicMax(&sMax[col], f32_to_key(lmax));
    }
    __syncthreads();
    if (tid < 128) {
        atomicAdd(&vsum[tid], sSum[tid]);
        atomicMax(&vmaxk[tid], sMax[tid]);
    }
}

// ------------------------------------------------------------- global MLPs
__global__ __launch_bounds__(128) void global_kernel(
    const float* __restrict__ vsum, const int* __restrict__ vmaxk,
    const float* __restrict__ z, const float* __restrict__ u,
    const float* __restrict__ Wg1, const float* __restrict__ bg1,
    const float* __restrict__ Wg2, const float* __restrict__ bg2,
    const float* __restrict__ Wu1, const float* __restrict__ bu1,
    const float* __restrict__ Wu2, const float* __restrict__ bu2,
    float* __restrict__ out_u, float* __restrict__ out_z)
{
    __shared__ float vmean[128], vmax[128], sh1[128], szp[128], sh2[128];
    const int t = threadIdx.x;

    vmean[t] = vsum[t] * (1.0f / (float)N_V);
    vmax[t]  = key_to_f32(vmaxk[t]);
    __syncthreads();

    // z' = relu([vmean, z] @ Wg1 + bg1) @ Wg2 + bg2
    float a = bg1[t];
    for (int k = 0; k < 128; ++k) a += vmean[k] * Wg1[k * 128 + t];
    for (int k = 0; k < 128; ++k) a += z[k] * Wg1[(128 + k) * 128 + t];
    sh1[t] = fmaxf(a, 0.0f);
    __syncthreads();
    float az = bg2[t];
    for (int k = 0; k < 128; ++k) az += sh1[k] * Wg2[k * 128 + t];
    szp[t] = az;
    out_z[t] = az;
    __syncthreads();

    // u' = relu([vmean, vmax, z', u] @ Wu1 + bu1) @ Wu2 + bu2
    float a2 = bu1[t];
    for (int k = 0; k < 128; ++k) a2 += vmean[k] * Wu1[k * 128 + t];
    for (int k = 0; k < 128; ++k) a2 += vmax[k] * Wu1[(128 + k) * 128 + t];
    for (int k = 0; k < 128; ++k) a2 += szp[k] * Wu1[(256 + k) * 128 + t];
    for (int k = 0; k < 64; ++k)  a2 += u[k] * Wu1[(384 + k) * 128 + t];
    sh2[t] = fmaxf(a2, 0.0f);
    __syncthreads();
    if (t < 64) {
        float au = bu2[t];
        for (int k = 0; k < 128; ++k) au += sh2[k] * Wu2[k * 64 + t];
        out_u[t] = au;
    }
}

// ------------------------------------------------------------------ launch
extern "C" void kernel_launch(void* const* d_in, const int* in_sizes, int n_in,
                              void* d_out, int out_size, void* d_ws, size_t ws_size,
                              hipStream_t stream) {
    const float* V   = (const float*)d_in[0];
    const float* E   = (const float*)d_in[1];
    const int*   ei  = (const int*)d_in[2];
    const float* u   = (const float*)d_in[3];
    const float* z   = (const float*)d_in[4];
    const float* We1 = (const float*)d_in[5];  const float* be1 = (const float*)d_in[6];
    const float* We2 = (const float*)d_in[7];  const float* be2 = (const float*)d_in[8];
    const float* Wn1 = (const float*)d_in[9];  const float* bn1 = (const float*)d_in[10];
    const float* Wn2 = (const float*)d_in[11]; const float* bn2 = (const float*)d_in[12];
    const float* Wg1 = (const float*)d_in[13]; const float* bg1 = (const float*)d_in[14];
    const float* Wg2 = (const float*)d_in[15]; const float* bg2 = (const float*)d_in[16];
    const float* Wu1 = (const float*)d_in[17]; const float* bu1 = (const float*)d_in[18];
    const float* Wu2 = (const float*)d_in[19]; const float* bu2 = (const float*)d_in[20];

    char* ws = (char*)d_ws;
    size_t off = 0;
    auto take = [&](size_t bytes) -> char* {
        off = (off + 255) & ~(size_t)255;
        char* p = ws + off;
        off += bytes;
        return p;
    };
    float*  edge_agg = (float*)take((size_t)N_V * 128 * 4);
    float*  deg      = (float*)take((size_t)N_V * 4);
    float*  vsum     = (float*)take(128 * 4);
    int*    vmaxk    = (int*)take(128 * 4);
    bf16_t* We1p     = (bf16_t*)take((size_t)320 * 128 * 2);
    bf16_t* We2p     = (bf16_t*)take((size_t)128 * 128 * 2);
    bf16_t* Wn1p     = (bf16_t*)take((size_t)448 * 128 * 2);
    bf16_t* Wn2p     = (bf16_t*)take((size_t)128 * 128 * 2);

    init_ws_kernel<<<(N_V * 128 + 255) / 256, 256, 0, stream>>>(edge_agg, deg, vsum, vmaxk);

    pack_w_kernel<<<(320 * 128 + 255) / 256, 256, 0, stream>>>(We1, We1p, 320, 128);
    pack_w_kernel<<<(128 * 128 + 255) / 256, 256, 0, stream>>>(We2, We2p, 128, 128);
    pack_w_kernel<<<(448 * 128 + 255) / 256, 256, 0, stream>>>(Wn1, Wn1p, 448, 128);
    pack_w_kernel<<<(128 * 128 + 255) / 256, 256, 0, stream>>>(Wn2, Wn2p, 128, 128);

    edge_kernel<<<N_E / 64, 256, 0, stream>>>(E, V, ei, ei + N_E, We1p, be1, We2p, be2,
                                              edge_agg, deg);

    node_kernel<<<(N_V + 63) / 64, 256, 0, stream>>>(V, edge_agg, deg, z, u,
                                                     Wn1p, bn1, Wn2p, bn2,
                                                     (float*)d_out, vsum, vmaxk);

    global_kernel<<<1, 128, 0, stream>>>(vsum, vmaxk, z, u,
                                         Wg1, bg1, Wg2, bg2, Wu1, bu1, Wu2, bu2,
                                         (float*)d_out + (size_t)N_V * 128,
                                         (float*)d_out + (size_t)N_V * 128 + 64);
}
